// MODEL_13194139533363
// MI455X (gfx1250) — matrix operand order repaired and checked by reading, not compiled, not run on hardware
//
#include <hip/hip_runtime.h>

// ---------------------------------------------------------------------------
// DKVMN forward for MI455X (gfx1250, wave32, WMMA).
// B=512, S=512, M=64, DK=128, DV=256, DF=128. N = B*S = 262144 rows.
// - All GEMMs hoisted out of the sequential scan, run as bf16
//   v_wmma_f32_16x16x32_bf16 (f32 accumulate).
// - erase/add GEMMs fused: each A fragment feeds two WMMA accumulators.
// - Scan: register-resident 64xf32 state per thread, software-pipelined
//   loads (double-buffered w in LDS, next-step erase/add in registers).
// ---------------------------------------------------------------------------

#define BB   512
#define SS   512
#define MM   64
#define DK   128
#define DV   256
#define DF   128
#define NROW (BB * SS)          // 262144

typedef __attribute__((ext_vector_type(16))) __bf16 v16bf;
typedef __attribute__((ext_vector_type(8)))  __bf16 v8bf;
typedef __attribute__((ext_vector_type(8)))  float  v8f;

// ---------------- WMMA fragment helpers (layouts per ISA 7.12.2) -----------

// A-matrix 16x32 bf16, row-major source [*, lda].
// lane 0-15 : row = row0+lane,    K = {k0..k0+7, k0+16..k0+23}
// lane 16-31: row = row0+lane-16, K = {k0+8..k0+15, k0+24..k0+31}
__device__ __forceinline__ v16bf load_a_frag(const __bf16* A, int lda,
                                             int row0, int k0, int lane) {
  int r  = row0 + (lane & 15);
  int kb = k0 + ((lane & 16) ? 8 : 0);
  const __bf16* p = A + (size_t)r * lda + kb;
  v8bf c0 = *(const v8bf*)(p);
  v8bf c1 = *(const v8bf*)(p + 16);
  v16bf f;
#pragma unroll
  for (int i = 0; i < 8; ++i) { f[i] = c0[i]; f[i + 8] = c1[i]; }
  return f;
}

// B-matrix 32x16 bf16 from weight stored [Ncols, K] row-major (x @ W^T form).
// lane 0-15 : col n = n0+lane,    K = {k0..k0+7, k0+16..k0+23}
// lane 16-31: col n = n0+lane-16, K = {k0+8..k0+15, k0+24..k0+31}
__device__ __forceinline__ v16bf load_b_frag(const __bf16* Bw, int ldb,
                                             int n0, int k0, int lane) {
  int n  = n0 + (lane & 15);
  int kb = k0 + ((lane & 16) ? 8 : 0);
  const __bf16* p = Bw + (size_t)n * ldb + kb;
  v8bf c0 = *(const v8bf*)(p);
  v8bf c1 = *(const v8bf*)(p + 16);
  v16bf f;
#pragma unroll
  for (int i = 0; i < 8; ++i) { f[i] = c0[i]; f[i + 8] = c1[i]; }
  return f;
}

__device__ __forceinline__ float sigmoidf_(float x) {
  return 1.0f / (1.0f + __expf(-x));
}

// D tile (v8f): element r holds (row0 + r + (lane>=16 ? 8:0), col0 + (lane&15))
template <int ACT>   // 0 = none, 1 = sigmoid, 2 = tanh
__device__ __forceinline__ void store_d_bf16(__bf16* C, int ldc, int row0,
                                             int col0, int lane, v8f d,
                                             const float* bias) {
  int n  = col0 + (lane & 15);
  float bi = bias ? bias[n] : 0.0f;
  int rb = row0 + ((lane & 16) ? 8 : 0);
#pragma unroll
  for (int r = 0; r < 8; ++r) {
    float v = d[r] + bi;
    if (ACT == 1) v = sigmoidf_(v);
    if (ACT == 2) v = tanhf(v);
    C[(size_t)(rb + r) * ldc + n] = (__bf16)v;
  }
}

// ---------------- small utility kernels ------------------------------------

__global__ void f2bf_kernel(const float* __restrict__ in,
                            __bf16* __restrict__ out, int n) {
  int i = blockIdx.x * 256 + threadIdx.x;
  if (i < n) out[i] = (__bf16)in[i];
}

// Embedding gather -> bf16 rows. One block per (b,s) row, 256 threads.
__global__ void gather_kernel(const int* __restrict__ qd,
                              const int* __restrict__ qad,
                              const float* __restrict__ qw,
                              const float* __restrict__ qaw,
                              __bf16* __restrict__ qe,
                              __bf16* __restrict__ qae) {
  int row = blockIdx.x;
  int t   = threadIdx.x;
  int qi  = qd[row];
  int qai = qad[row];
  if (t < DK) qe[(size_t)row * DK + t] = (__bf16)qw[(size_t)qi * DK + t];
  qae[(size_t)row * DV + t] = (__bf16)qaw[(size_t)qai * DV + t];
}

// ---------------- scores + softmax -> attention weights w ------------------
// Block = 128 threads (4 waves). Block computes 16 rows x 64 cols of
// q_e @ key_mem^T (K=128) with WMMA, then row-softmax over M=64 in LDS.
__global__ __launch_bounds__(128)
void scores_softmax_kernel(const __bf16* __restrict__ qe,
                           const __bf16* __restrict__ keyb,
                           float* __restrict__ wout) {
  int lane = threadIdx.x & 31;
  int wv   = threadIdx.x >> 5;          // 0..3
  int row0 = blockIdx.x * 16;
  int col0 = wv * 16;

  v8f acc = {};
#pragma unroll
  for (int kc = 0; kc < DK; kc += 32) {
    v16bf a = load_a_frag(qe, DK, row0, kc, lane);
    v16bf b = load_b_frag(keyb, DK, col0, kc, lane);
    acc = __builtin_amdgcn_wmma_f32_16x16x32_bf16(false, a, false, b,
                                                  (short)0, acc, false, false);
  }

  __shared__ float st[16][MM + 4];
  {
    int n  = col0 + (lane & 15);
    int rb = (lane & 16) ? 8 : 0;
#pragma unroll
    for (int r = 0; r < 8; ++r) st[rb + r][n] = acc[r];
  }
  __syncthreads();

  if (threadIdx.x < 16) {
    int r = threadIdx.x;
    float mx = -3.0e38f;
#pragma unroll 8
    for (int j = 0; j < MM; ++j) mx = fmaxf(mx, st[r][j]);
    float s = 0.0f;
#pragma unroll 8
    for (int j = 0; j < MM; ++j) s += __expf(st[r][j] - mx);
    float inv = 1.0f / s;
    float* wr = wout + (size_t)(row0 + r) * MM;
#pragma unroll 8
    for (int j = 0; j < MM; ++j) wr[j] = __expf(st[r][j] - mx) * inv;
  }
}

// ---------------- generic bf16 WMMA GEMM with fused activation -------------
// C[N, ldc] = act(A[N, K] @ Bw^T + bias), tile 32x64 per 256-thread block.
template <int ACT, int K>
__global__ __launch_bounds__(256)
void gemm_bf16_kernel(const __bf16* __restrict__ A, int lda,
                      const __bf16* __restrict__ Bw, int ldb,
                      const float* __restrict__ bias,
                      __bf16* __restrict__ C, int ldc) {
  int lane = threadIdx.x & 31;
  int wv   = threadIdx.x >> 5;                       // 0..7
  int row0 = blockIdx.x * 32 + (wv >> 2) * 16;
  int col0 = blockIdx.y * 64 + (wv & 3) * 16;

  v8f acc = {};
#pragma unroll
  for (int kc = 0; kc < K; kc += 32) {
    v16bf a = load_a_frag(A, lda, row0, kc, lane);
    v16bf b = load_b_frag(Bw, ldb, col0, kc, lane);
    acc = __builtin_amdgcn_wmma_f32_16x16x32_bf16(false, a, false, b,
                                                  (short)0, acc, false, false);
  }
  store_d_bf16<ACT>(C, ldc, row0, col0, lane, acc, bias);
}

// ---------------- fused erase+add GEMM -------------------------------------
// Both GEMMs share the same A (qa_e); load each A fragment once and feed two
// accumulators (2 WMMA per A fragment -> half the A traffic of two launches).
__global__ __launch_bounds__(256)
void gemm_erase_add_kernel(const __bf16* __restrict__ qae,
                           const __bf16* __restrict__ ew,   // [DV,DV]
                           const __bf16* __restrict__ aw,   // [DV,DV]
                           const float* __restrict__ ebias,
                           const float* __restrict__ abias,
                           __bf16* __restrict__ er,
                           __bf16* __restrict__ ad) {
  int lane = threadIdx.x & 31;
  int wv   = threadIdx.x >> 5;
  int row0 = blockIdx.x * 32 + (wv >> 2) * 16;
  int col0 = blockIdx.y * 64 + (wv & 3) * 16;

  v8f acce = {};
  v8f acca = {};
#pragma unroll
  for (int kc = 0; kc < DV; kc += 32) {
    v16bf a  = load_a_frag(qae, DV, row0, kc, lane);
    v16bf b0 = load_b_frag(ew, DV, col0, kc, lane);
    v16bf b1 = load_b_frag(aw, DV, col0, kc, lane);
    acce = __builtin_amdgcn_wmma_f32_16x16x32_bf16(false, a, false, b0,
                                                   (short)0, acce, false, false);
    acca = __builtin_amdgcn_wmma_f32_16x16x32_bf16(false, a, false, b1,
                                                   (short)0, acca, false, false);
  }
  store_d_bf16<1>(er, DV, row0, col0, lane, acce, ebias);
  store_d_bf16<2>(ad, DV, row0, col0, lane, acca, abias);
}

// ---------------- rce GEMM: A = concat(reads[:,0:256], ie[:,0:128]) --------
__global__ __launch_bounds__(256)
void gemm_rce_kernel(const __bf16* __restrict__ reads,
                     const __bf16* __restrict__ ie,
                     const __bf16* __restrict__ rw,   // [DF, DV+DF]
                     const float* __restrict__ bias,
                     __bf16* __restrict__ rce) {
  int lane = threadIdx.x & 31;
  int wv   = threadIdx.x >> 5;
  int row0 = blockIdx.x * 32 + (wv >> 2) * 16;
  int col0 = blockIdx.y * 64 + (wv & 3) * 16;

  v8f acc = {};
#pragma unroll
  for (int c = 0; c < (DV + DF) / 32; ++c) {
    v16bf a = (c < DV / 32)
                  ? load_a_frag(reads, DV, row0, c * 32, lane)
                  : load_a_frag(ie, DF, row0, (c - DV / 32) * 32, lane);
    v16bf b = load_b_frag(rw, DV + DF, col0, c * 32, lane);
    acc = __builtin_amdgcn_wmma_f32_16x16x32_bf16(false, a, false, b,
                                                  (short)0, acc, false, false);
  }
  store_d_bf16<2>(rce, DF, row0, col0, lane, acc, bias);
}

// ---------------- sequential memory scan -----------------------------------
// One block per batch b. Thread d (0..255) owns DV column d; the 64-row Mv
// column lives entirely in registers. Software-pipelined: step t+1's w row
// (double-buffered LDS) and erase/add (registers) are prefetched while step
// t's 64-FMA chain executes. One barrier per step.
__global__ __launch_bounds__(256)
void scan_kernel(const float* __restrict__ wall,     // [B,S,M]
                 const __bf16* __restrict__ er,      // [B,S,DV]
                 const __bf16* __restrict__ ad,      // [B,S,DV]
                 const float* __restrict__ initM,    // [M,DV]
                 __bf16* __restrict__ reads) {       // [B,S,DV]
  int b = blockIdx.x;
  int d = threadIdx.x;

  float Mv[MM];
#pragma unroll
  for (int m = 0; m < MM; ++m) Mv[m] = initM[(size_t)m * DV + d];

  __shared__ float ws[2][MM];
  const float*  wb = wall  + (size_t)b * SS * MM;
  const __bf16* eb = er    + (size_t)b * SS * DV;
  const __bf16* ab = ad    + (size_t)b * SS * DV;
  __bf16*       rb = reads + (size_t)b * SS * DV;

  // prologue: stage step 0
  if (d < MM) ws[0][d] = wb[d];
  float e = (float)eb[d];
  float a = (float)ab[d];
  __syncthreads();

  for (int t = 0; t < SS; ++t) {
    int cur = t & 1;
    // prefetch step t+1 while step t computes
    float en = 0.0f, an = 0.0f;
    if (t + 1 < SS) {
      if (d < MM) ws[cur ^ 1][d] = wb[(size_t)(t + 1) * MM + d];
      en = (float)eb[(size_t)(t + 1) * DV + d];
      an = (float)ab[(size_t)(t + 1) * DV + d];
    }

    float racc = 0.0f;
#pragma unroll
    for (int m = 0; m < MM; ++m) {
      float wm = ws[cur][m];
      racc = fmaf(wm, Mv[m], racc);               // read before update
      Mv[m] = fmaf(Mv[m], 1.0f - wm * e, wm * a); // Mv*(1-w*e) + w*a
    }
    rb[(size_t)t * DV + d] = (__bf16)racc;

    e = en;
    a = an;
    __syncthreads();   // separates step t's ws reads from step t+2's writes
  }
}

// ---------------- prediction head + masked BCE -----------------------------

__global__ void zero_accum_kernel(float* a) {
  if (threadIdx.x < 2) a[threadIdx.x] = 0.0f;
}

__global__ __launch_bounds__(256)
void pred_loss_kernel(const __bf16* __restrict__ rce,
                      const float* __restrict__ pw,   // [DF]
                      const float* __restrict__ pb,   // [1]
                      const float* __restrict__ target,
                      float* __restrict__ out,        // [3N+1]
                      float* __restrict__ accum) {
  int row = blockIdx.x * 256 + threadIdx.x;
  const __bf16* r = rce + (size_t)row * DF;
  float dotv = 0.0f;
#pragma unroll
  for (int k = 0; k < DF; k += 8) {
    v8bf c = *(const v8bf*)(r + k);
#pragma unroll
    for (int i = 0; i < 8; ++i) dotv = fmaf((float)c[i], pw[k + i], dotv);
  }
  float pred = dotv + pb[0];

  out[row] = sigmoidf_(pred);                                // pred_logit
  float t   = target[row];
  bool mask = (t >= 0.0f);
  float bce = fmaxf(pred, 0.0f) - pred * t + log1pf(__expf(-fabsf(pred)));
  out[NROW + 1 + row]     = sigmoidf_(mask ? pred : 0.0f);   // filtered_pred
  out[2 * NROW + 1 + row] = mask ? t : 0.0f;                 // filtered_target

  __shared__ float sb[256], sc[256];
  sb[threadIdx.x] = mask ? bce : 0.0f;
  sc[threadIdx.x] = mask ? 1.0f : 0.0f;
  __syncthreads();
  for (int s = 128; s > 0; s >>= 1) {
    if (threadIdx.x < s) {
      sb[threadIdx.x] += sb[threadIdx.x + s];
      sc[threadIdx.x] += sc[threadIdx.x + s];
    }
    __syncthreads();
  }
  if (threadIdx.x == 0) {
    atomicAdd(&accum[0], sb[0]);
    atomicAdd(&accum[1], sc[0]);
  }
}

__global__ void finalize_loss_kernel(const float* __restrict__ accum,
                                     float* __restrict__ out) {
  out[NROW] = accum[0] / fmaxf(accum[1], 1.0f);
}

// ---------------- host-side launch -----------------------------------------

extern "C" void kernel_launch(void* const* d_in, const int* in_sizes, int n_in,
                              void* d_out, int out_size, void* d_ws,
                              size_t ws_size, hipStream_t stream) {
  const int*   q_data   = (const int*)d_in[0];
  const int*   qa_data  = (const int*)d_in[1];
  const float* target   = (const float*)d_in[2];
  const float* q_emb    = (const float*)d_in[3];
  const float* qa_emb   = (const float*)d_in[4];
  const float* key_mem  = (const float*)d_in[5];
  const float* init_mem = (const float*)d_in[6];
  const float* erase_w  = (const float*)d_in[7];
  const float* erase_b  = (const float*)d_in[8];
  const float* add_w    = (const float*)d_in[9];
  const float* add_b    = (const float*)d_in[10];
  const float* input_w  = (const float*)d_in[11];
  const float* input_b  = (const float*)d_in[12];
  const float* read_w   = (const float*)d_in[13];
  const float* read_b   = (const float*)d_in[14];
  const float* pred_w   = (const float*)d_in[15];
  const float* pred_b   = (const float*)d_in[16];
  float* out = (float*)d_out;

  char* ws = (char*)d_ws;
  size_t off = 0;
  auto take = [&](size_t bytes) {
    char* p = ws + off;
    off += (bytes + 255) & ~(size_t)255;
    return p;
  };
  __bf16* qe_b   = (__bf16*)take((size_t)NROW * DK * 2);
  __bf16* qae_b  = (__bf16*)take((size_t)NROW * DV * 2);
  float*  w_f    = (float*) take((size_t)NROW * MM * 4);
  __bf16* er_b   = (__bf16*)take((size_t)NROW * DV * 2);
  __bf16* ad_b   = (__bf16*)take((size_t)NROW * DV * 2);
  __bf16* rd_b   = (__bf16*)take((size_t)NROW * DV * 2);
  __bf16* ie_b   = (__bf16*)take((size_t)NROW * DF * 2);
  __bf16* rce_b  = (__bf16*)take((size_t)NROW * DF * 2);
  __bf16* key_bf = (__bf16*)take((size_t)MM * DK * 2);
  __bf16* ew_bf  = (__bf16*)take((size_t)DV * DV * 2);
  __bf16* aw_bf  = (__bf16*)take((size_t)DV * DV * 2);
  __bf16* iw_bf  = (__bf16*)take((size_t)DF * DK * 2);
  __bf16* rw_bf  = (__bf16*)take((size_t)DF * (DV + DF) * 2);
  float*  accum  = (float*) take(2 * sizeof(float));

  // 1) weights -> bf16
  auto cvt = [&](const float* src, __bf16* dst, int n) {
    f2bf_kernel<<<(n + 255) / 256, 256, 0, stream>>>(src, dst, n);
  };
  cvt(key_mem, key_bf, MM * DK);
  cvt(erase_w, ew_bf, DV * DV);
  cvt(add_w,   aw_bf, DV * DV);
  cvt(input_w, iw_bf, DF * DK);
  cvt(read_w,  rw_bf, DF * (DV + DF));

  // 2) embedding gather -> bf16 activations
  gather_kernel<<<NROW, 256, 0, stream>>>(q_data, qa_data, q_emb, qa_emb,
                                          qe_b, qae_b);

  // 3) w = softmax(q_e @ key_mem^T)   [N, 64]
  scores_softmax_kernel<<<NROW / 16, 128, 0, stream>>>(qe_b, key_bf, w_f);

  // 4) erase = sigmoid(qa_e @ erase_w^T), add = tanh(qa_e @ add_w^T), fused
  {
    dim3 g(NROW / 32, DV / 64);
    gemm_erase_add_kernel<<<g, 256, 0, stream>>>(qae_b, ew_bf, aw_bf,
                                                 erase_b, add_b, er_b, ad_b);
  }

  // 5) ie = tanh(q_e @ input_w^T)     [N, 128]
  {
    dim3 g(NROW / 32, DF / 64);
    gemm_bf16_kernel<2, DK><<<g, 256, 0, stream>>>(qe_b, DK, iw_bf, DK,
                                                   input_b, ie_b, DF);
  }

  // 6) sequential scan -> reads       [B,S,DV]
  scan_kernel<<<BB, 256, 0, stream>>>(w_f, er_b, ad_b, init_mem, rd_b);

  // 7) rce = tanh(concat(reads, ie) @ read_w^T)  [N, 128]
  {
    dim3 g(NROW / 32, DF / 64);
    gemm_rce_kernel<<<g, 256, 0, stream>>>(rd_b, ie_b, rw_bf, read_b, rce_b);
  }

  // 8) prediction head, masked BCE, outputs
  zero_accum_kernel<<<1, 32, 0, stream>>>(accum);
  pred_loss_kernel<<<NROW / 256, 256, 0, stream>>>(rce_b, pred_w, pred_b,
                                                   target, out, accum);
  finalize_loss_kernel<<<1, 1, 0, stream>>>(accum, out);
}
